// Encoder_8014408974536
// MI455X (gfx1250) — matrix operand order repaired and checked by reading, not compiled, not run on hardware
//
#include <hip/hip_runtime.h>
#include <math.h>
#include <stdint.h>

// ---------------------------------------------------------------------------
// Transformer encoder forward (B=4, S=512, D=512, H=8, DK=DV=64, DFF=2048, L=6)
// CDNA5 / gfx1250: wave32, WMMA f32_16x16x32_f16 (f16 in, f32 accumulate).
//  - GEMM: 8 waves/block, block tile 64x128, 2x2 WMMA register tiling/wave.
//  - Activations q/k/v/ff kept in f16 in global (producer GEMM converts once)
//    -> halves re-read traffic, removes v_cvt chains in consumers.
//  - Attention: fused scores->mask->softmax->prob write->ctx; V tile staged
//    to LDS with GLOBAL_LOAD_ASYNC_TO_LDS_B128 (ASYNCcnt path).
// ---------------------------------------------------------------------------

typedef _Float16 v16h __attribute__((ext_vector_type(16)));
typedef float    v8f  __attribute__((ext_vector_type(8)));

#define NB      4      // batch
#define SEQ     512    // sequence length
#define DMODEL  512
#define NHEAD   8
#define DHEAD   64
#define DFF     2048
#define NLAYER  6
#define NTOK    (NB * SEQ)          // 2048 rows
#define HD      (NHEAD * DHEAD)     // 512

__device__ __forceinline__ v8f wmma_f16(v16h a, v16h b, v8f c) {
    return __builtin_amdgcn_wmma_f32_16x16x32_f16(false, a, false, b,
                                                  (short)0, c, false, false);
}

__device__ __forceinline__ float gelu_f(float x) {
    const float c = 0.7978845608028654f; // sqrt(2/pi)
    float t = c * (x + 0.044715f * x * x * x);
    return 0.5f * x * (1.0f + tanhf(t));
}

// K offset inside a 32-wide A fragment slab for register pair i (contiguous: k = 16*hi + 2*i).
__device__ __forceinline__ int afrag_k(int i, int kb) {
    return kb + 2 * i;
}

// ---------------------------------------------------------------------------
// Embedding + positional encoding: h[b,s,:] = emb[x[b,s]] + pe[s]
// ---------------------------------------------------------------------------
__global__ __launch_bounds__(256) void embed_kernel(
    const int* __restrict__ x, const float* __restrict__ emb,
    const float* __restrict__ pe, float* __restrict__ h)
{
    const int bs  = blockIdx.x;          // 0..NTOK-1
    const int s   = bs & (SEQ - 1);
    const int tok = x[bs];
    const float* e = emb + (size_t)tok * DMODEL;
    const float* p = pe  + (size_t)s   * DMODEL;
    float* o = h + (size_t)bs * DMODEL;
    for (int d = threadIdx.x; d < DMODEL; d += 256)
        o[d] = e[d] + p[d];
}

// ---------------------------------------------------------------------------
// Generic GEMM: C[M,N] = act(A[M,K] @ B[K,N] + bias) (+ residual)
// 256 threads = 8 waves laid out 2(M) x 4(N); each wave owns a 32x32 C tile
// (2x2 WMMA register tiling -> 4 v_wmma per 32-wide K step).
// Block tile: 64(M) x 128(N).
// ACT: 0 none / 1 tanh-GELU.  AF16: A matrix in f16.  OUTF16: f16 output.
// ---------------------------------------------------------------------------
template<int ACT, int AF16, int OUTF16>
__global__ __launch_bounds__(256) void gemm_bias_kernel(
    const void* __restrict__ Ain, const float* __restrict__ Bw,
    const float* __restrict__ bias, const float* __restrict__ residual,
    void* __restrict__ Cout, int M, int N, int K)
{
    __shared__ __align__(16) _Float16 As[64][40];    // 64x32 A tile, padded
    __shared__ __align__(16) _Float16 Bs[32][136];   // 32x128 B tile, padded

    const int tid  = threadIdx.x;
    const int wave = tid >> 5;
    const int lane = tid & 31;
    const int lm   = lane & 15;
    const int hi   = lane >> 4;            // 0: lanes 0-15, 1: lanes 16-31
    const int m0   = blockIdx.y * 64;
    const int n0   = blockIdx.x * 128;
    const int wm   = (wave >> 2) << 5;     // wave row tile in block (0/32)
    const int wn   = (wave & 3)  << 5;     // wave col tile in block (0..96)
    const int kbA  = hi * 16;
    const int kbB  = hi * 16;

    v8f acc00 = {}, acc01 = {}, acc10 = {}, acc11 = {};

    for (int k0 = 0; k0 < K; k0 += 32) {
        // Stage A: 64 rows x 32 cols.
        if (AF16) {
            // f16 source: straight 16-byte copies, no conversion.
            const _Float16* Af = (const _Float16*)Ain;
            const int row = tid >> 2;          // 0..63
            const int ch  = (tid & 3) << 3;    // 0,8,16,24
            const uint4 u = *(const uint4*)(
                Af + (size_t)(m0 + row) * K + k0 + ch);
            *(uint4*)&As[row][ch] = u;
        } else {
            const int row = tid >> 3;          // 0..31
            const int col = (tid & 7) << 2;    // 0..28
            const float* Af = (const float*)Ain;
            #pragma unroll
            for (int rr = 0; rr < 2; ++rr) {
                const float4 f = *(const float4*)(
                    Af + (size_t)(m0 + row + rr * 32) * K + k0 + col);
                As[row + rr * 32][col + 0] = (_Float16)f.x;
                As[row + rr * 32][col + 1] = (_Float16)f.y;
                As[row + rr * 32][col + 2] = (_Float16)f.z;
                As[row + rr * 32][col + 3] = (_Float16)f.w;
            }
        }
        // Stage B: 32 rows x 128 cols (f32 weights -> f16).
        {
            const int row = tid >> 5;          // 0..7
            const int col = lane << 2;         // 0..124
            #pragma unroll
            for (int rr = 0; rr < 4; ++rr) {
                const float4 f = *(const float4*)(
                    Bw + (size_t)(k0 + row + rr * 8) * N + n0 + col);
                Bs[row + rr * 8][col + 0] = (_Float16)f.x;
                Bs[row + rr * 8][col + 1] = (_Float16)f.y;
                Bs[row + rr * 8][col + 2] = (_Float16)f.z;
                Bs[row + rr * 8][col + 3] = (_Float16)f.w;
            }
        }
        __syncthreads();

        v16h a0, a1, b0, b1;
        #pragma unroll
        for (int i = 0; i < 8; ++i) {
            const int kk = afrag_k(i, kbA);
            a0[2 * i]     = As[wm + lm][kk];
            a0[2 * i + 1] = As[wm + lm][kk + 1];
            a1[2 * i]     = As[wm + 16 + lm][kk];
            a1[2 * i + 1] = As[wm + 16 + lm][kk + 1];
        }
        #pragma unroll
        for (int j = 0; j < 16; ++j) {
            b0[j] = Bs[kbB + j][wn + lm];
            b1[j] = Bs[kbB + j][wn + 16 + lm];
        }

        acc00 = wmma_f16(a0, b0, acc00);
        acc01 = wmma_f16(a0, b1, acc01);
        acc10 = wmma_f16(a1, b0, acc10);
        acc11 = wmma_f16(a1, b1, acc11);
        __syncthreads();
    }

    // Epilogue: bias (+ GELU) (+ residual), f32 or f16 store.
    const float bn0 = bias ? bias[n0 + wn + lm]      : 0.0f;
    const float bn1 = bias ? bias[n0 + wn + 16 + lm] : 0.0f;

    auto store_tile = [&](const v8f& acc, int ms, int ns, float bn) {
        const int gcol = n0 + wn + ns * 16 + lm;
        #pragma unroll
        for (int r = 0; r < 8; ++r) {
            const int grow = m0 + wm + ms * 16 + r + hi * 8;
            float val = acc[r] + bn;
            if (ACT == 1) val = gelu_f(val);
            if (residual) val += residual[(size_t)grow * N + gcol];
            if (OUTF16)
                ((_Float16*)Cout)[(size_t)grow * N + gcol] = (_Float16)val;
            else
                ((float*)Cout)[(size_t)grow * N + gcol] = val;
        }
    };
    store_tile(acc00, 0, 0, bn0);
    store_tile(acc01, 0, 1, bn1);
    store_tile(acc10, 1, 0, bn0);
    store_tile(acc11, 1, 1, bn1);
}

// ---------------------------------------------------------------------------
// LayerNorm over last dim (D=512). One block (256 thr) per token row.
// ---------------------------------------------------------------------------
__global__ __launch_bounds__(256) void ln_kernel(
    const float* __restrict__ X, const float* __restrict__ g,
    const float* __restrict__ b, float* __restrict__ Y)
{
    __shared__ float red[256];
    const int row = blockIdx.x;
    const int tid = threadIdx.x;
    const float* xr = X + (size_t)row * DMODEL;
    float x0 = xr[tid];
    float x1 = xr[tid + 256];

    red[tid] = x0 + x1;
    __syncthreads();
    #pragma unroll
    for (int s = 128; s > 0; s >>= 1) {
        if (tid < s) red[tid] += red[tid + s];
        __syncthreads();
    }
    const float mu = red[0] * (1.0f / DMODEL);
    __syncthreads();

    const float d0 = x0 - mu, d1 = x1 - mu;
    red[tid] = d0 * d0 + d1 * d1;
    __syncthreads();
    #pragma unroll
    for (int s = 128; s > 0; s >>= 1) {
        if (tid < s) red[tid] += red[tid + s];
        __syncthreads();
    }
    const float inv = rsqrtf(red[0] * (1.0f / DMODEL) + 1e-5f);

    float* yr = Y + (size_t)row * DMODEL;
    yr[tid]       = d0 * inv * g[tid]       + b[tid];
    yr[tid + 256] = d1 * inv * g[tid + 256] + b[tid + 256];
}

// ---------------------------------------------------------------------------
// Fused attention for one (batch, head, 16-query tile). q/k/v are f16.
//   scores (WMMA) -> key-pad mask -> softmax (LDS) -> prob write -> ctx (WMMA)
// Block: 256 thr = 8 waves. Grid: (SEQ/16, NHEAD, NB).
// Phase 3: split-K across wave pairs; V tiles staged to LDS with
// GLOBAL_LOAD_ASYNC_TO_LDS_B128 (ASYNCcnt), combined via LDS partials.
// ---------------------------------------------------------------------------
__global__ __launch_bounds__(256) void attn_kernel(
    const _Float16* __restrict__ q, const _Float16* __restrict__ k,
    const _Float16* __restrict__ v, const int* __restrict__ x,
    float* __restrict__ ctx, float* __restrict__ attn_out, int layer)
{
    __shared__ float    sc[16][SEQ];              // 16 q rows x 512 keys
    __shared__ __align__(16) _Float16 Vs[64][80]; // 64 keys x 64 dims, 160B row
    __shared__ float    red[16][16];
    __shared__ float    rowstat[16];
    __shared__ float    pacc[4][16][16];

    const int qt   = blockIdx.x;      // query tile (0..31)
    const int hh   = blockIdx.y;      // head
    const int bb   = blockIdx.z;      // batch
    const int tid  = threadIdx.x;
    const int wave = tid >> 5;
    const int lane = tid & 31;
    const int lm   = lane & 15;
    const int hi   = lane >> 4;
    const int kbA  = hi * 16;
    const int kbB  = hi * 16;

    // --- Phase 1: scores = Q K^T * scale, masked --------------------------
    v16h a0, a1;
    {
        const _Float16* qp =
            q + ((size_t)bb * SEQ + qt * 16 + lm) * HD + hh * DHEAD;
        #pragma unroll
        for (int i = 0; i < 8; ++i) {
            const int kk = afrag_k(i, kbA);
            a0[2 * i]     = qp[kk];
            a0[2 * i + 1] = qp[kk + 1];
            a1[2 * i]     = qp[32 + kk];
            a1[2 * i + 1] = qp[32 + kk + 1];
        }
    }
    const float scale = 0.125f;       // 1/sqrt(DK)
    #pragma unroll
    for (int t = 0; t < 4; ++t) {
        const int kt     = wave * 4 + t;      // key tile 0..31
        const int keycol = kt * 16 + lm;
        const _Float16* kp = k + ((size_t)bb * SEQ + keycol) * HD
                               + hh * DHEAD + hi * 16;
        v8f  acc = {};
        v16h bf;
        #pragma unroll
        for (int j = 0; j < 16; ++j) bf[j] = kp[j];
        acc = wmma_f16(a0, bf, acc);
        #pragma unroll
        for (int j = 0; j < 16; ++j) bf[j] = kp[32 + j];
        acc = wmma_f16(a1, bf, acc);

        const bool masked = (x[bb * SEQ + keycol] == 0);  // PAD == 0
        #pragma unroll
        for (int r = 0; r < 8; ++r)
            sc[r + hi * 8][keycol] = masked ? -1e9f : acc[r] * scale;
    }
    __syncthreads();

    // --- Phase 2: softmax over 512 keys; write probabilities --------------
    const int row = tid >> 4;         // 0..15
    const int seg = tid & 15;         // 0..15, each owns 32 contiguous keys
    float lmax = -3.4e38f;
    #pragma unroll
    for (int c = 0; c < 32; ++c)
        lmax = fmaxf(lmax, sc[row][seg * 32 + c]);
    red[row][seg] = lmax;
    __syncthreads();
    if (seg == 0) {
        float m = red[row][0];
        #pragma unroll
        for (int j = 1; j < 16; ++j) m = fmaxf(m, red[row][j]);
        rowstat[row] = m;
    }
    __syncthreads();
    const float rmax = rowstat[row];
    float lsum = 0.0f;
    #pragma unroll
    for (int c = 0; c < 32; ++c) {
        const float e = __expf(sc[row][seg * 32 + c] - rmax);
        sc[row][seg * 32 + c] = e;
        lsum += e;
    }
    red[row][seg] = lsum;
    __syncthreads();
    if (seg == 0) {
        float s = 0.0f;
        #pragma unroll
        for (int j = 0; j < 16; ++j) s += red[row][j];
        rowstat[row] = 1.0f / s;
    }
    __syncthreads();
    const float inv = rowstat[row];
    float* ao = attn_out +
        ((((size_t)bb * NLAYER + layer) * NHEAD + hh) * SEQ + qt * 16 + row)
        * (size_t)SEQ;
    #pragma unroll
    for (int c = 0; c < 32; ++c) {
        const float p = sc[row][seg * 32 + c] * inv;
        sc[row][seg * 32 + c] = p;
        ao[seg * 32 + c] = p;
    }
    __syncthreads();

    // --- Phase 3: ctx = P @ V  (16 x 512 @ 512 x 64) ----------------------
    // nt = wave&3 -> 16-wide DV slice; khalf = wave>>2 -> key-range half.
    const int nt    = wave & 3;
    const int khalf = wave >> 2;
    const uint32_t vsbase = (uint32_t)(uintptr_t)(&Vs[0][0]);
    v8f acc = {};
    for (int kk0 = 0; kk0 < SEQ; kk0 += 64) {
        // Async-copy V[kk0..kk0+63][hh*64..+63] (f16) straight into LDS:
        // 16 B per lane per op, ASYNCcnt-tracked, no VGPR round-trip.
        {
            const int r0 = tid >> 3;           // 0..31
            const int ch = (tid & 7) << 3;     // half index 0..56
            #pragma unroll
            for (int rr = 0; rr < 64; rr += 32) {
                const uint32_t ldsa =
                    vsbase + (uint32_t)(((r0 + rr) * 80 + ch) * 2);
                const _Float16* gp =
                    v + ((size_t)bb * SEQ + kk0 + r0 + rr) * HD
                      + hh * DHEAD + ch;
                asm volatile("global_load_async_to_lds_b128 %0, %1, off"
                             :: "v"(ldsa), "v"(gp) : "memory");
            }
        }
        asm volatile("s_wait_asynccnt 0x0" ::: "memory");
        __syncthreads();

        v16h a, bf;
        const int kk = kk0 + khalf * 32;
        #pragma unroll
        for (int i = 0; i < 8; ++i) {
            const int kidx = kk + afrag_k(i, kbA);
            a[2 * i]     = (_Float16)sc[lm][kidx];
            a[2 * i + 1] = (_Float16)sc[lm][kidx + 1];
        }
        #pragma unroll
        for (int j = 0; j < 16; ++j)
            bf[j] = Vs[khalf * 32 + kbB + j][nt * 16 + lm];
        acc = wmma_f16(a, bf, acc);
        __syncthreads();
    }
    if (khalf == 1) {
        #pragma unroll
        for (int r = 0; r < 8; ++r)
            pacc[nt][r + hi * 8][lm] = acc[r];
    }
    __syncthreads();
    if (khalf == 0) {
        #pragma unroll
        for (int r = 0; r < 8; ++r) {
            const int srow = qt * 16 + r + hi * 8;
            ctx[((size_t)bb * SEQ + srow) * HD + hh * DHEAD + nt * 16 + lm]
                = acc[r] + pacc[nt][r + hi * 8][lm];
        }
    }
}

// ---------------------------------------------------------------------------
__global__ __launch_bounds__(256) void copy_kernel(
    const float* __restrict__ src, float* __restrict__ dst, int n)
{
    for (int i = blockIdx.x * 256 + threadIdx.x; i < n; i += gridDim.x * 256)
        dst[i] = src[i];
}

// ---------------------------------------------------------------------------
extern "C" void kernel_launch(void* const* d_in, const int* in_sizes, int n_in,
                              void* d_out, int out_size, void* d_ws,
                              size_t ws_size, hipStream_t stream)
{
    (void)in_sizes; (void)n_in; (void)out_size; (void)ws_size;

    const int*   x     = (const int*)  d_in[0];
    const float* emb   = (const float*)d_in[1];
    const float* pe    = (const float*)d_in[2];
    const float* WQ    = (const float*)d_in[3];
    const float* bQ    = (const float*)d_in[4];
    const float* WK    = (const float*)d_in[5];
    const float* bK    = (const float*)d_in[6];
    const float* WV    = (const float*)d_in[7];
    const float* bV    = (const float*)d_in[8];
    const float* WO    = (const float*)d_in[9];
    const float* bO    = (const float*)d_in[10];
    const float* ln1g  = (const float*)d_in[11];
    const float* ln1b  = (const float*)d_in[12];
    const float* W1    = (const float*)d_in[13];
    const float* b1    = (const float*)d_in[14];
    const float* W2    = (const float*)d_in[15];
    const float* b2    = (const float*)d_in[16];
    const float* ln2g  = (const float*)d_in[17];
    const float* ln2b  = (const float*)d_in[18];

    const size_t TOKD = (size_t)NTOK * DMODEL;    // 1,048,576 elements
    char* base = (char*)d_ws;
    float*    h   = (float*)   (base);                         // f32 [NTOK,D]
    float*    tmp = (float*)   (base + 4  * TOKD);             // f32 [NTOK,D]
    float*    cb  = (float*)   (base + 8  * TOKD);             // f32 ctx
    _Float16* qb  = (_Float16*)(base + 12 * TOKD);             // f16 [NTOK,HD]
    _Float16* kb  = (_Float16*)(base + 14 * TOKD);
    _Float16* vb  = (_Float16*)(base + 16 * TOKD);
    _Float16* ff  = (_Float16*)(base + 18 * TOKD);             // f16 [NTOK,DFF]

    float* out_h    = (float*)d_out;
    float* out_attn = out_h + TOKD;

    embed_kernel<<<NTOK, 256, 0, stream>>>(x, emb, pe, h);

    const dim3 blk(256);
    const dim3 g512(DMODEL / 128, NTOK / 64);  // N=512  -> (4, 32)
    const dim3 gff (DFF    / 128, NTOK / 64);  // N=2048 -> (16, 32)
    const dim3 gattn(SEQ / 16, NHEAD, NB);

    for (int l = 0; l < NLAYER; ++l) {
        const float* WQl = WQ + (size_t)l * DMODEL * HD;
        const float* WKl = WK + (size_t)l * DMODEL * HD;
        const float* WVl = WV + (size_t)l * DMODEL * HD;
        const float* WOl = WO + (size_t)l * HD * DMODEL;
        const float* W1l = W1 + (size_t)l * DMODEL * DFF;
        const float* W2l = W2 + (size_t)l * DFF * DMODEL;

        // QKV projections: f32 A (h), f16 outputs.
        gemm_bias_kernel<0, 0, 1><<<g512, blk, 0, stream>>>(
            h, WQl, bQ + (size_t)l * HD, nullptr, qb, NTOK, HD, DMODEL);
        gemm_bias_kernel<0, 0, 1><<<g512, blk, 0, stream>>>(
            h, WKl, bK + (size_t)l * HD, nullptr, kb, NTOK, HD, DMODEL);
        gemm_bias_kernel<0, 0, 1><<<g512, blk, 0, stream>>>(
            h, WVl, bV + (size_t)l * HD, nullptr, vb, NTOK, HD, DMODEL);

        attn_kernel<<<gattn, blk, 0, stream>>>(qb, kb, vb, x, cb, out_attn, l);

        // O projection + residual (f32 in/out).
        gemm_bias_kernel<0, 0, 0><<<g512, blk, 0, stream>>>(
            cb, WOl, bO + (size_t)l * DMODEL, h, tmp, NTOK, DMODEL, HD);
        ln_kernel<<<NTOK, blk, 0, stream>>>(
            tmp, ln1g + (size_t)l * DMODEL, ln1b + (size_t)l * DMODEL, h);

        // FFN1: GELU, f16 output.  FFN2: f16 A, residual, f32 output.
        gemm_bias_kernel<1, 0, 1><<<gff, blk, 0, stream>>>(
            h, W1l, b1 + (size_t)l * DFF, nullptr, ff, NTOK, DFF, DMODEL);
        gemm_bias_kernel<0, 1, 0><<<g512, blk, 0, stream>>>(
            ff, W2l, b2 + (size_t)l * DMODEL, h, tmp, NTOK, DMODEL, DFF);
        ln_kernel<<<NTOK, blk, 0, stream>>>(
            tmp, ln2g + (size_t)l * DMODEL, ln2b + (size_t)l * DMODEL, h);
    }

    copy_kernel<<<512, blk, 0, stream>>>(h, out_h, (int)TOKD);
}
